// PCdown_76888504533072
// MI455X (gfx1250) — matrix operand order repaired and checked by reading, not compiled, not run on hardware
//
#include <hip/hip_runtime.h>
#include <hip/hip_bf16.h>

typedef __attribute__((ext_vector_type(16))) _Float16 v16h;
typedef __attribute__((ext_vector_type(8)))  float    v8f;

// ---------------- problem constants ----------------
#define PB 4
#define PN 8192
#define PS 4096      // N/2
#define PK 16        // knn k
#define MIDC 32
#define OUTC 64
#define KPAD 96      // 67 padded to 3 K-tiles of 32
#define EPS 1e-5f

// ---------------- tiny elementwise kernels ----------------

// pre1[p, c] = b1[c] + dot(x[p,:3], W1[c,:3])   (t over B*N*32)
__global__ void k_mlp1(const float* __restrict__ x, const float* __restrict__ W1,
                       const float* __restrict__ b1, float* __restrict__ pre1, int total) {
    int t = blockIdx.x * blockDim.x + threadIdx.x;
    if (t >= total) return;
    int c = t & (MIDC - 1);
    int p = t >> 5;
    const float* xp = x + (size_t)p * 3;
    const float* wp = W1 + (size_t)c * 3;
    pre1[t] = b1[c] + xp[0] * wp[0] + xp[1] * wp[1] + xp[2] * wp[2];
}

__global__ void k_zero_stats(float* __restrict__ s) {
    if (threadIdx.x < 128) s[threadIdx.x] = 0.0f;
}

// column sums & sumsq; X is [rows, C] row-major, C in {32,64}, C | 256.
// stats[0..C-1]=sum, stats[64..64+C-1]=sumsq
__global__ void k_col_stats(const float* __restrict__ X, int total, int C,
                            float* __restrict__ stats) {
    __shared__ float ss[256], sq[256];
    int tid = threadIdx.x;
    float s = 0.f, q = 0.f;
    for (long i = (long)blockIdx.x * 256 + tid; i < total; i += (long)gridDim.x * 256) {
        float v = X[i];
        s += v; q += v * v;
    }
    ss[tid] = s; sq[tid] = q;
    __syncthreads();
    if (tid < C) {
        float a = 0.f, b = 0.f;
        for (int j = tid; j < 256; j += C) { a += ss[j]; b += sq[j]; }
        atomicAdd(&stats[tid], a);
        atomicAdd(&stats[64 + tid], b);
    }
}

// stats[128+c]=scale, stats[192+c]=shift: y = scale*x + shift == gamma*(x-mean)*rsqrt(var+eps)+beta
__global__ void k_finalize_stats(const float* __restrict__ g, const float* __restrict__ be,
                                 float invRows, int C, float* __restrict__ stats) {
    int c = threadIdx.x;
    if (c < C) {
        float mean = stats[c] * invRows;
        float var  = stats[64 + c] * invRows - mean * mean;
        float sc   = g[c] * rsqrtf(var + EPS);
        stats[128 + c] = sc;
        stats[192 + c] = be[c] - mean * sc;
    }
}

// y16 = (f16) relu(scale*x + shift)
__global__ void k_bnrelu_f16(const float* __restrict__ X, const float* __restrict__ stats,
                             int total, int Cmask, _Float16* __restrict__ Y) {
    int i = blockIdx.x * blockDim.x + threadIdx.x;
    if (i >= total) return;
    int c = i & Cmask;
    float v = fmaf(X[i], stats[128 + c], stats[192 + c]);
    Y[i] = (_Float16)fmaxf(v, 0.0f);
}

// Pack weight matrix W[64, Ksrc] (f32) into WMMA B-fragment order (f16, K padded to 'K'):
//   out[((tn*nkt + kt)*32 + lane)*16 + e] = W[tn*16+lane%16][kt*32 + (e<8 ? 8*(lane/16)+e : 16+8*(lane/16)+e-8)]
// so each lane's 16 halves are 32 contiguous bytes -> 2x global_load_b128 in the GEMM.
__global__ void k_pack_b(const float* __restrict__ W, int Ksrc, int K,
                         _Float16* __restrict__ out) {
    int t = blockIdx.x * blockDim.x + threadIdx.x;
    if (t >= OUTC * K) return;
    int f = t >> 9;         // fragment id = tn*nkt + kt
    int r = t & 511;
    int l = r >> 4;         // lane: n = lane%16, half = lane/16
    int e = r & 15;         // elem: k in table order
    int nkt = K >> 5;
    int tn = f / nkt, kt = f - tn * nkt;
    int n = tn * 16 + (l & 15);
    int k = kt * 32 + ((e < 8) ? (8 * (l >> 4) + e) : (16 + 8 * (l >> 4) + (e - 8)));
    out[t] = (_Float16)((k < Ksrc) ? W[(size_t)n * Ksrc + k] : 0.0f);
}

// ---------------- WMMA GEMM:  C[M,64] = A[M,K](f16) * W^T + bias ----------------
// one wave computes a 16x64 strip: one A fragment feeds 4 WMMAs per K-tile.
// Bpack is fragment-major (see k_pack_b). K multiple of 32, M multiple of 16.
__global__ void __launch_bounds__(32)
k_wmma_gemm_bias(const _Float16* __restrict__ A, const _Float16* __restrict__ Bpack,
                 const float* __restrict__ bias, float* __restrict__ C,
                 int M, int K) {
    int tm = blockIdx.x;
    int lane = threadIdx.x & 31;
    int lrow = lane & 15;
    int half = lane >> 4;

    int m = tm * 16 + lrow;
    const unsigned* Arow = (const unsigned*)(A + (size_t)m * K);  // rows 4B-aligned
    int nkt = K >> 5;
    const uint4* bbase = (const uint4*)Bpack + lane * 2;  // 512 halves = 64 uint4 per fragment
    size_t tstride = (size_t)nkt * 64;                    // uint4 stride between N-tiles

    v8f acc0 = {}, acc1 = {}, acc2 = {}, acc3 = {};
    for (int kt = 0; kt < nkt; ++kt) {
        // A fragment (16x32 f16): lane=M, dword p -> K pair per ISA layout (two b128 runs)
        union { v16h v; unsigned u[8]; } av;
        int kb = kt * 16;
#pragma unroll
        for (int p = 0; p < 8; ++p) {
            int dw = (p < 4) ? (4 * half + p) : (8 + 4 * half + (p - 4));
            av.u[p] = Arow[kb + dw];
        }
        // B fragments for the 4 N-tiles, each lane 32 contiguous bytes
        const uint4* bq = bbase + (size_t)kt * 64;
        union { v16h v; uint4 q[2]; } b0, b1, b2, b3;
        b0.q[0] = bq[0];            b0.q[1] = bq[1];
        b1.q[0] = bq[tstride];      b1.q[1] = bq[tstride + 1];
        b2.q[0] = bq[2 * tstride];  b2.q[1] = bq[2 * tstride + 1];
        b3.q[0] = bq[3 * tstride];  b3.q[1] = bq[3 * tstride + 1];

        acc0 = __builtin_amdgcn_wmma_f32_16x16x32_f16(false, av.v, false, b0.v, (short)0, acc0, false, false);
        acc1 = __builtin_amdgcn_wmma_f32_16x16x32_f16(false, av.v, false, b1.v, (short)0, acc1, false, false);
        acc2 = __builtin_amdgcn_wmma_f32_16x16x32_f16(false, av.v, false, b2.v, (short)0, acc2, false, false);
        acc3 = __builtin_amdgcn_wmma_f32_16x16x32_f16(false, av.v, false, b3.v, (short)0, acc3, false, false);
    }

    int mr0 = tm * 16 + half * 8;
#define WG_EPILOGUE(T, ACC)                                                   \
    {                                                                         \
        int n = (T) * 16 + lrow;                                              \
        float bb = bias[n];                                                   \
        _Pragma("unroll") for (int r = 0; r < 8; ++r)                         \
            C[(size_t)(mr0 + r) * OUTC + n] = ACC[r] + bb;                    \
    }
    WG_EPILOGUE(0, acc0)
    WG_EPILOGUE(1, acc1)
    WG_EPILOGUE(2, acc2)
    WG_EPILOGUE(3, acc3)
#undef WG_EPILOGUE
}

// ---------------- FPS: one 1024-thread workgroup per batch; cloud+dist in 320KB LDS ----------------
__global__ void __launch_bounds__(1024)
k_fps(const float* __restrict__ xin, int* __restrict__ fidx) {
    extern __shared__ float sm[];                       // 4 * 8192 floats = 128KB dynamic LDS
    float* sx = sm;
    float* sy = sm + PN;
    float* sz = sm + 2 * PN;
    float* sd = sm + 3 * PN;
    __shared__ float rv[1024];
    __shared__ int   ri[1024];
    __shared__ int   sfar;

    int b = blockIdx.x, tid = threadIdx.x;
    const float* base = xin + (size_t)b * PN * 3;
    for (int i = tid; i < PN; i += 1024) {
        sx[i] = base[i * 3 + 0];
        sy[i] = base[i * 3 + 1];
        sz[i] = base[i * 3 + 2];
        sd[i] = 1e10f;
    }
    if (tid == 0) sfar = 0;
    __syncthreads();

    for (int it = 0; it < PS; ++it) {
        int far = sfar;
        if (tid == 0) fidx[b * PS + it] = far;
        float cx = sx[far], cy = sy[far], cz = sz[far];
        float bv = -1.0f; int bi = 0;
        for (int i = tid; i < PN; i += 1024) {
            float dx = sx[i] - cx, dy = sy[i] - cy, dz = sz[i] - cz;
            float d = dx * dx + dy * dy + dz * dz;
            float nd = fminf(sd[i], d);
            sd[i] = nd;
            if (nd > bv) { bv = nd; bi = i; }
        }
        rv[tid] = bv; ri[tid] = bi;
        __syncthreads();
        for (int s2 = 512; s2 > 0; s2 >>= 1) {
            if (tid < s2) {
                float v2 = rv[tid + s2]; int i2 = ri[tid + s2];
                if (v2 > rv[tid] || (v2 == rv[tid] && i2 < ri[tid])) { rv[tid] = v2; ri[tid] = i2; }
            }
            __syncthreads();
        }
        if (tid == 0) sfar = ri[0];
        __syncthreads();
    }
}

// ---------------- kNN: thread-per-query, register top-16, LDS-staged ref tiles ----------------
#define KCH 2048
__global__ void __launch_bounds__(128)
k_knn(const float* __restrict__ xin, const int* __restrict__ fidx, int* __restrict__ kidx) {
    __shared__ float sx[KCH], sy[KCH], sz[KCH];
    int b = blockIdx.y;
    int s = blockIdx.x * 128 + threadIdx.x;
    const float* base = xin + (size_t)b * PN * 3;
    int qi = fidx[b * PS + s];
    float qx = base[qi * 3], qy = base[qi * 3 + 1], qz = base[qi * 3 + 2];

    float bd[PK]; int bi[PK];
#pragma unroll
    for (int j = 0; j < PK; ++j) { bd[j] = 3e38f; bi[j] = 0; }

    for (int c0 = 0; c0 < PN; c0 += KCH) {
        __syncthreads();
        for (int j = threadIdx.x; j < KCH; j += 128) {
            sx[j] = base[(c0 + j) * 3 + 0];
            sy[j] = base[(c0 + j) * 3 + 1];
            sz[j] = base[(c0 + j) * 3 + 2];
        }
        __syncthreads();
        // overlap: prefetch next chunk into cache while scanning this one
        if (c0 + KCH < PN)
            __builtin_prefetch(base + (size_t)(c0 + KCH + threadIdx.x * 16) * 3, 0, 0);
        for (int j = 0; j < KCH; ++j) {
            float dx = sx[j] - qx, dy = sy[j] - qy, dz = sz[j] - qz;
            float d = dx * dx + dy * dy + dz * dz;
            if (d < bd[PK - 1]) {
                bd[PK - 1] = d; bi[PK - 1] = c0 + j;
#pragma unroll
                for (int t = PK - 1; t > 0; --t) {
                    if (bd[t] < bd[t - 1]) {
                        float tv = bd[t]; bd[t] = bd[t - 1]; bd[t - 1] = tv;
                        int ti = bi[t]; bi[t] = bi[t - 1]; bi[t - 1] = ti;
                    }
                }
            }
        }
    }
    int* outp = kidx + ((size_t)b * PS + s) * PK;
#pragma unroll
    for (int j = 0; j < PK; ++j) outp[j] = bi[j];
}

// ---------------- gather + max-pool over K, emit f16 rows padded to 96 ----------------
__global__ void __launch_bounds__(64)
k_group_pool(const float* __restrict__ xin, const float* __restrict__ feat,
             const int* __restrict__ fidx, const int* __restrict__ kidx,
             _Float16* __restrict__ pooled) {
    __shared__ int sidx[PK];
    __shared__ float cc[3];
    int q = blockIdx.x;                 // b*PS + s
    int b = q >> 12;                    // PS = 4096
    int tid = threadIdx.x;
    if (tid < PK) sidx[tid] = kidx[(size_t)q * PK + tid];
    if (tid == 0) {
        int qi = fidx[q];
        const float* p = xin + ((size_t)b * PN + qi) * 3;
        cc[0] = p[0]; cc[1] = p[1]; cc[2] = p[2];
    }
    __syncthreads();

    _Float16* orow = pooled + (size_t)q * KPAD;

    // feature channel tid: max over 16 neighbors
    float fm = -3e38f;
    for (int k = 0; k < PK; ++k) {
        int id = sidx[k];
        fm = fmaxf(fm, feat[((size_t)b * PN + id) * OUTC + tid]);
    }
    orow[3 + tid] = (_Float16)fm;

    if (tid < 3) {  // relative-xyz channels
        float ctr = cc[tid];
        float mm = -3e38f;
        for (int k = 0; k < PK; ++k) {
            int id = sidx[k];
            mm = fmaxf(mm, xin[((size_t)b * PN + id) * 3 + tid] - ctr);
        }
        orow[tid] = (_Float16)mm;
    }
    if (tid < KPAD - 67) orow[67 + tid] = (_Float16)0.0f;   // zero pad 67..95
}

// ---------------- orchestration ----------------
extern "C" void kernel_launch(void* const* d_in, const int* in_sizes, int n_in,
                              void* d_out, int out_size, void* d_ws, size_t ws_size,
                              hipStream_t stream) {
    (void)in_sizes; (void)n_in; (void)out_size; (void)ws_size;
    const float* x_in = (const float*)d_in[0];
    const float* W1   = (const float*)d_in[1];
    const float* b1   = (const float*)d_in[2];
    const float* g1   = (const float*)d_in[3];
    const float* be1  = (const float*)d_in[4];
    const float* W2   = (const float*)d_in[5];
    const float* b2   = (const float*)d_in[6];
    const float* W3   = (const float*)d_in[7];
    const float* b3   = (const float*)d_in[8];
    const float* g2   = (const float*)d_in[9];
    const float* be2  = (const float*)d_in[10];
    const float* W4   = (const float*)d_in[11];
    const float* b4   = (const float*)d_in[12];
    float* out = (float*)d_out;

    char* ws = (char*)d_ws;
    float*    pre1   = (float*)   (ws + 0);          // 4 MB  [B*N, 32]
    float*    feat   = (float*)   (ws + 4194304);    // 8 MB  [B*N, 64]
    _Float16* h16    = (_Float16*)(ws + 12582912);   // 2 MB  [B*N, 32]
    int*      fidx   = (int*)     (ws + 14680064);   // 64 KB [B*S]
    int*      kidx   = (int*)     (ws + 14745600);   // 1 MB  [B*S, 16]
    _Float16* pooled = (_Float16*)(ws + 15794176);   // 3 MB  [B*S, 96]
    float*    pre3   = (float*)   (ws + 18939904);   // 4 MB  [B*S, 64]
    _Float16* x16    = (_Float16*)(ws + 23134208);   // 2 MB  [B*S, 64]
    _Float16* w2p    = (_Float16*)(ws + 25231360);   // packed fragments
    _Float16* w3p    = (_Float16*)(ws + 25235456);
    _Float16* w4p    = (_Float16*)(ws + 25247744);
    float*    stats  = (float*)   (ws + 25255936);   // 256 floats

    const int nPts   = PB * PN;          // 32768
    const int tot1   = nPts * MIDC;      // 1,048,576
    const int nQ     = PB * PS;          // 16384
    const int tot3   = nQ * OUTC;        // 1,048,576

    // weights -> packed f16 WMMA fragments (W3 zero-padded to K=96)
    k_pack_b<<<(OUTC * MIDC + 255) / 256, 256, 0, stream>>>(W2, MIDC, MIDC, w2p);
    k_pack_b<<<(OUTC * KPAD + 255) / 256, 256, 0, stream>>>(W3, 67, KPAD, w3p);
    k_pack_b<<<(OUTC * OUTC + 255) / 256, 256, 0, stream>>>(W4, OUTC, OUTC, w4p);

    // stage 1 MLP + BN stats + ReLU -> f16
    k_mlp1<<<tot1 / 256, 256, 0, stream>>>(x_in, W1, b1, pre1, tot1);
    k_zero_stats<<<1, 128, 0, stream>>>(stats);
    k_col_stats<<<256, 256, 0, stream>>>(pre1, tot1, MIDC, stats);
    k_finalize_stats<<<1, 64, 0, stream>>>(g1, be1, 1.0f / (float)nPts, MIDC, stats);
    k_bnrelu_f16<<<tot1 / 256, 256, 0, stream>>>(pre1, stats, tot1, MIDC - 1, h16);

    // feat = h16 @ W2^T + b2   (WMMA, 16x64 strip per wave)
    k_wmma_gemm_bias<<<nPts / 16, 32, 0, stream>>>(h16, w2p, b2, feat, nPts, MIDC);

    // FPS (LDS-resident cloud, 128KB dynamic LDS per workgroup)
    k_fps<<<PB, 1024, 4 * PN * sizeof(float), stream>>>(x_in, fidx);

    // kNN
    {
        dim3 g(PS / 128, PB);
        k_knn<<<g, 128, 0, stream>>>(x_in, fidx, kidx);
    }

    // gather + max pool -> [B*S, 96] f16
    k_group_pool<<<nQ, 64, 0, stream>>>(x_in, feat, fidx, kidx, pooled);

    // pre3 = pooled @ W3p^T + b3   (WMMA, K=96)
    k_wmma_gemm_bias<<<nQ / 16, 32, 0, stream>>>(pooled, w3p, b3, pre3, nQ, KPAD);

    // BN2 + ReLU -> f16
    k_zero_stats<<<1, 128, 0, stream>>>(stats);
    k_col_stats<<<256, 256, 0, stream>>>(pre3, tot3, OUTC, stats);
    k_finalize_stats<<<1, 64, 0, stream>>>(g2, be2, 1.0f / (float)nQ, OUTC, stats);
    k_bnrelu_f16<<<tot3 / 256, 256, 0, stream>>>(pre3, stats, tot3, OUTC - 1, x16);

    // out = x16 @ W4^T + b4   (WMMA)
    k_wmma_gemm_bias<<<nQ / 16, 32, 0, stream>>>(x16, w4p, b4, out, nQ, OUTC);
}
